// Algin_38603166056827
// MI455X (gfx1250) — compile-verified, matrix operand order checked by reading, not run on hardware
//
#include <hip/hip_runtime.h>
#include <stdint.h>
#include <math.h>

// ---------------------------------------------------------------------------
// CDNA5 (gfx1250) deformable-alignment pipeline: im2col + bf16 WMMA GEMMs.
// Every conv / deform-conv channel contraction runs on v_wmma_f32_16x16x32_bf16.
// GEMM: weight panel staged in LDS once per block, hoisted to registers once
// per wave, then each wave streams MT=4 M-tiles with A-only global traffic.
// ---------------------------------------------------------------------------

typedef __attribute__((ext_vector_type(16))) __bf16 v16bf;
typedef __attribute__((ext_vector_type(8)))  float  v8f;

union BF16x16 { uint32_t u[8]; v16bf v; };

__device__ __forceinline__ uint32_t f2bf(float f) {
  uint32_t u = __builtin_bit_cast(uint32_t, f);
  u += 0x7FFFu + ((u >> 16) & 1u);          // round-to-nearest-even
  return (u >> 16) & 0xFFFFu;
}

// ---------------------------------------------------------------------------
// Weight prepack: OIHW f32 -> per-lane bf16 B-operand panels.
// Layout: wp[((ntile*nk + kstep)*32 + lane)*8 + v] holds K-pair (2 bf16).
// lane<16 : N=lane,     K = ks*32 + {0..7} (v0..3) / {16..23} (v4..7)
// lane>=16: N=lane-16,  K = ks*32 + {8..15}        / {24..31}
// ---------------------------------------------------------------------------
__global__ void prepack_w(const float* __restrict__ w, uint32_t* __restrict__ wp,
                          int Cout, int Kreal, int nk) {
  int lane  = threadIdx.x & 31;
  int v     = threadIdx.x >> 5;     // 0..7
  int ntile = blockIdx.x;
  int ks    = blockIdx.y;
  int o  = ntile * 16 + (lane & 15);
  int k0 = ks * 32 + ((v & 3) * 2) + ((v & 4) ? 16 : 0) + ((lane >= 16) ? 8 : 0);
  uint32_t lo = 0, hi = 0;
  if (o < Cout) {
    if (k0     < Kreal) lo = f2bf(w[(size_t)o * Kreal + k0]);
    if (k0 + 1 < Kreal) hi = f2bf(w[(size_t)o * Kreal + k0 + 1]);
  }
  wp[(((size_t)ntile * nk + ks) * 32 + lane) * 8 + v] = lo | (hi << 16);
}

// ---------------------------------------------------------------------------
// im2col (3x3, stride 1) with optional 2-tensor channel concat, f32 -> bf16.
// A panel: row-major [pixel][KP] bf16, K index = cin*9 + (ky*3+kx), zero-padded.
// Templated on KP/KREAL so all index divisors are compile-time constants.
// ---------------------------------------------------------------------------
template <int KP, int KREAL>
__global__ void im2col_bf16(const float* __restrict__ in0, const float* __restrict__ in1,
                            uint32_t* __restrict__ A, int hi, int wi,
                            int ho, int wo, int pad, uint32_t total) {
  uint32_t idx = blockIdx.x * blockDim.x + threadIdx.x;
  if (idx >= total) return;
  constexpr uint32_t kw = KP >> 1;
  uint32_t pix = idx / kw;                 // constant divisor -> magic multiply
  uint32_t kp  = idx - pix * kw;
  uint32_t plane = (uint32_t)ho * wo;
  uint32_t b   = pix / plane;
  uint32_t rem = pix - b * plane;
  int y = (int)(rem / (uint32_t)wo);
  int x = (int)(rem - (uint32_t)y * wo);
  uint32_t packed = 0;
#pragma unroll
  for (int half = 0; half < 2; ++half) {
    uint32_t k = kp * 2 + half;
    float val = 0.f;
    if (k < (uint32_t)KREAL) {
      uint32_t cin = k / 9u, tap = k - cin * 9u;
      int yi = y + (int)(tap / 3u) - pad;
      int xi = x + (int)(tap % 3u) - pad;
      if (yi >= 0 && yi < hi && xi >= 0 && xi < wi) {
        const float* src = (cin < 16u) ? in0 : in1;
        uint32_t c = cin & 15u;
        val = src[(((size_t)b * 16 + c) * hi + yi) * wi + xi];
      }
    }
    packed |= f2bf(val) << (16 * half);
  }
  A[idx] = packed;
}

// ---------------------------------------------------------------------------
// Deformable im2col: bilinear sample unr features at offset taps, -> bf16 panel.
// K index = c*9 + t ; DG=8 groups of Cg=2 channels ; offsets [B,144,h,w]; KP=160.
// ---------------------------------------------------------------------------
__global__ void deform_im2col_bf16(const float* __restrict__ xf, const float* __restrict__ off,
                                   uint32_t* __restrict__ A, int h, int w, uint32_t total) {
  uint32_t idx = blockIdx.x * blockDim.x + threadIdx.x;
  if (idx >= total) return;
  constexpr uint32_t kw = 80;              // KP=160
  uint32_t pix = idx / kw;
  uint32_t kp  = idx - pix * kw;
  uint32_t plane = (uint32_t)h * w;
  uint32_t b   = pix / plane;
  uint32_t rem = pix - b * plane;
  int y = (int)(rem / (uint32_t)w);
  int x = (int)(rem - (uint32_t)y * w);
  uint32_t packed = 0;
#pragma unroll
  for (int half = 0; half < 2; ++half) {
    uint32_t k = kp * 2 + half;
    float val = 0.f;
    if (k < 144u) {
      uint32_t c = k / 9u, t = k - c * 9u, g = c >> 1;
      size_t obase = ((size_t)b * 144 + (size_t)(g * 9u + t) * 2) * plane
                   + (size_t)y * w + x;
      float oy = off[obase];
      float ox = off[obase + plane];
      float sy = (float)(y + (int)(t / 3u) - 1) + oy;
      float sx = (float)(x + (int)(t % 3u) - 1) + ox;
      float fy = floorf(sy), fx = floorf(sx);
      float wy = sy - fy,    wx = sx - fx;
      int y0 = (int)fy, x0 = (int)fx;
      const float* xp = xf + ((size_t)b * 16 + c) * plane;
      auto g4 = [&](int yy, int xx) -> float {
        return (yy >= 0 && yy < h && xx >= 0 && xx < w) ? xp[(size_t)yy * w + xx] : 0.f;
      };
      float v00 = g4(y0, x0),     v01 = g4(y0, x0 + 1);
      float v10 = g4(y0 + 1, x0), v11 = g4(y0 + 1, x0 + 1);
      val = v00 * (1.f - wy) * (1.f - wx) + v01 * (1.f - wy) * wx
          + v10 * wy * (1.f - wx)         + v11 * wy * wx;
    }
    packed |= f2bf(val) << (16 * half);
  }
  A[idx] = packed;
}

// ---------------------------------------------------------------------------
// WMMA GEMM: D[P x Cout] = A[P x KP](bf16) * W[KP x Cout](bf16) + bias, opt ReLU.
// 8 waves/block; B panel staged via LDS -> registers once; each wave runs MT
// consecutive 16x16 tiles with only A-panel global traffic in steady state.
// NCHW epilogue uses one 32-bit udiv per tile (batch split) and no y/x split.
// ---------------------------------------------------------------------------
template <int NK, int MT>
__global__ void gemm_wmma_bf16(const uint32_t* __restrict__ A, const uint32_t* __restrict__ wp,
                               const float* __restrict__ bias, float* __restrict__ out,
                               uint32_t P, int Cout, uint32_t plane, int relu) {
  __shared__ uint32_t wsh[NK * 256];

  int lane = threadIdx.x & 31;
  int wave = threadIdx.x >> 5;
  int ntile = blockIdx.y;

  // cooperative stage of this N-tile's weight panel into LDS (ds_store path)
  const uint32_t* wg = wp + (size_t)ntile * NK * 256;
#pragma unroll
  for (int i = 0; i < NK; ++i)
    wsh[i * 256 + threadIdx.x] = wg[i * 256 + threadIdx.x];
  __syncthreads();

  // hoist per-lane B fragments LDS -> registers (ds_load_b128 x2 per step)
  BF16x16 bm[NK];
#pragma unroll
  for (int ks = 0; ks < NK; ++ks)
#pragma unroll
    for (int v = 0; v < 8; ++v)
      bm[ks].u[v] = wsh[ks * 256 + lane * 8 + v];

  int   koffLane = (lane >= 16) ? 4 : 0;   // dword offset of high-lane K half
  int   ncol  = lane & 15;
  int   nch   = ntile * 16 + ncol;
  float bs    = bias[nch];
  uint32_t rbase = (lane >= 16) ? 8u : 0u;

  uint32_t mtile0 = ((uint32_t)blockIdx.x * 8 + wave) * MT;
#pragma unroll 1
  for (int t = 0; t < MT; ++t) {
    uint32_t mtile = mtile0 + t;
    uint32_t mbase = mtile * 16;
    if (mbase >= P) break;                            // wave-uniform

    const uint32_t* Ap = A + (size_t)(mbase + (lane & 15)) * (NK * 16);

    v8f acc = {0.f, 0.f, 0.f, 0.f, 0.f, 0.f, 0.f, 0.f};
#pragma unroll
    for (int ks = 0; ks < NK; ++ks) {
      BF16x16 a;
      int base = ks * 16 + koffLane;                  // 32 K-values = 16 dwords/step
#pragma unroll
      for (int v = 0; v < 8; ++v)
        a.u[v] = Ap[base + (v & 3) + ((v & 4) ? 8 : 0)];
      if (ks + 1 < NK)
        __builtin_prefetch((const void*)(Ap + (ks + 1) * 16), 0, 1);
      acc = __builtin_amdgcn_wmma_f32_16x16x32_bf16(false, a.v, false, bm[ks].v,
                                                    (short)0, acc, false, false);
    }

    // D layout: lane<16 -> rows 0..7, lane>=16 -> rows 8..15 ; col = lane%16
    uint32_t bidx    = mbase / plane;                 // single 32-bit udiv per tile
    uint32_t rembase = mbase - bidx * plane + rbase;
#pragma unroll
    for (int r = 0; r < 8; ++r) {
      float val = acc[r] + bs;
      if (relu) val = fmaxf(val, 0.f);
      uint32_t b   = bidx;
      uint32_t rem = rembase + (uint32_t)r;
      if (rem >= plane) { b += 1; rem -= plane; }     // tiles cross batch at most once
      out[((size_t)b * Cout + nch) * plane + rem] = val;
    }
  }
}

// ---------------------------------------------------------------------------
// Host orchestration
// ---------------------------------------------------------------------------
static inline unsigned blocks_for(long n) { return (unsigned)((n + 255) / 256); }

extern "C" void kernel_launch(void* const* d_in, const int* in_sizes, int n_in,
                              void* d_out, int out_size, void* d_ws, size_t ws_size,
                              hipStream_t stream) {
  (void)in_sizes; (void)n_in; (void)out_size; (void)ws_size;
  const float* ref_img   = (const float*)d_in[0];
  const float* unr_img   = (const float*)d_in[1];
  const float* fe_w[3]   = {(const float*)d_in[2], (const float*)d_in[4], (const float*)d_in[6]};
  const float* fe_b[3]   = {(const float*)d_in[3], (const float*)d_in[5], (const float*)d_in[7]};
  const float* off_w[3]  = {(const float*)d_in[8], (const float*)d_in[10], (const float*)d_in[12]};
  const float* off_b[3]  = {(const float*)d_in[9], (const float*)d_in[11], (const float*)d_in[13]};
  const float* dcnoff_w  = (const float*)d_in[14];
  const float* dcnoff_b  = (const float*)d_in[15];
  const float* dcn_w     = (const float*)d_in[16];
  const float* dcn_b     = (const float*)d_in[17];
  float* outp = (float*)d_out;

  const int  hs[3] = {192, 190, 188};
  long Ps[3];
  for (int i = 0; i < 3; ++i) Ps[i] = 4L * hs[i] * hs[i];

  // ---- carve workspace ----
  char*  ws  = (char*)d_ws;
  size_t off = 0;
  auto alloc = [&](size_t bytes) -> void* {
    off = (off + 255) & ~(size_t)255;
    void* p = ws + off;
    off += bytes;
    return p;
  };
  float* fref[3]; float* funr[3];
  for (int i = 0; i < 3; ++i) fref[i] = (float*)alloc((size_t)Ps[i] * 16 * 4);
  for (int i = 0; i < 3; ++i) funr[i] = (float*)alloc((size_t)Ps[i] * 16 * 4);
  float*    offfeat = (float*)alloc((size_t)Ps[0] * 16 * 4);
  float*    offsets = (float*)alloc((size_t)Ps[0] * 144 * 4);
  uint32_t* Abuf    = (uint32_t*)alloc((size_t)Ps[0] * 288 * 2);   // shared im2col/sbuf panel
  uint32_t* fe_wp[3];
  for (int i = 0; i < 3; ++i) fe_wp[i] = (uint32_t*)alloc(5 * 256 * 4);
  uint32_t* off_wp[3];
  for (int i = 0; i < 3; ++i) off_wp[i] = (uint32_t*)alloc(9 * 256 * 4);
  uint32_t* dcnoff_wp = (uint32_t*)alloc(9 * 5 * 256 * 4);
  uint32_t* dcn_wp    = (uint32_t*)alloc(5 * 256 * 4);

  // ---- pack all weights ----
  for (int i = 0; i < 3; ++i)
    prepack_w<<<dim3(1, 5), 256, 0, stream>>>(fe_w[i],  fe_wp[i],  16, 144, 5);
  for (int i = 0; i < 3; ++i)
    prepack_w<<<dim3(1, 9), 256, 0, stream>>>(off_w[i], off_wp[i], 16, 288, 9);
  prepack_w<<<dim3(9, 5), 256, 0, stream>>>(dcnoff_w, dcnoff_wp, 144, 144, 5);
  prepack_w<<<dim3(1, 5), 256, 0, stream>>>(dcn_w,    dcn_wp,    16, 144, 5);

  const int MT = 4;
  auto gemm = [&](const uint32_t* A, const uint32_t* wp, const float* bias, float* o,
                  long P, int nk, int Cout, int h, int relu) {
    long mtiles = (P + 15) / 16;
    dim3 grid((unsigned)((mtiles + 8 * MT - 1) / (8 * MT)), (unsigned)(Cout / 16));
    uint32_t plane = (uint32_t)h * h;
    if (nk == 5)
      gemm_wmma_bf16<5, MT><<<grid, 256, 0, stream>>>(A, wp, bias, o, (uint32_t)P, Cout, plane, relu);
    else
      gemm_wmma_bf16<9, MT><<<grid, 256, 0, stream>>>(A, wp, bias, o, (uint32_t)P, Cout, plane, relu);
  };

  // ---- shared feature extraction (both images) ----
  for (int s = 0; s < 2; ++s) {
    const float* img = (s == 0) ? ref_img : unr_img;
    float** f = (s == 0) ? fref : funr;
    long t0 = Ps[0] * 80;   // KP/2 = 80
    im2col_bf16<160, 144><<<blocks_for(t0), 256, 0, stream>>>(
        img, img, Abuf, 192, 192, 192, 192, 1, (uint32_t)t0);
    gemm(Abuf, fe_wp[0], fe_b[0], f[0], Ps[0], 5, 16, 192, 1);
    long t1 = Ps[1] * 80;
    im2col_bf16<160, 144><<<blocks_for(t1), 256, 0, stream>>>(
        f[0], f[0], Abuf, 192, 192, 190, 190, 0, (uint32_t)t1);
    gemm(Abuf, fe_wp[1], fe_b[1], f[1], Ps[1], 5, 16, 190, 1);
    long t2 = Ps[2] * 80;
    im2col_bf16<160, 144><<<blocks_for(t2), 256, 0, stream>>>(
        f[1], f[1], Abuf, 190, 190, 188, 188, 0, (uint32_t)t2);
    gemm(Abuf, fe_wp[2], fe_b[2], f[2], Ps[2], 5, 16, 188, 1);
  }

  // ---- per-level deformable alignment (coarse-to-fine as in reference) ----
  long outOff[3] = {0, Ps[0] * 16, Ps[0] * 16 + Ps[1] * 16};
  for (int i = 2; i >= 0; --i) {
    int  h = hs[i];
    long P = Ps[i];
    // offset-feature conv: cat(ref,unr) 32ch, pad=1, K=288
    long tc = P * 144;
    im2col_bf16<288, 288><<<blocks_for(tc), 256, 0, stream>>>(
        fref[i], funr[i], Abuf, h, h, h, h, 1, (uint32_t)tc);
    gemm(Abuf, off_wp[i], off_b[i], offfeat, P, 9, 16, h, 1);
    // DCN offset prediction: 16 -> 144, pad=1
    long to = P * 80;
    im2col_bf16<160, 144><<<blocks_for(to), 256, 0, stream>>>(
        offfeat, offfeat, Abuf, h, h, h, h, 1, (uint32_t)to);
    gemm(Abuf, dcnoff_wp, dcnoff_b, offsets, P, 5, 144, h, 0);
    // deformable sampling -> bf16 panel
    long td = P * 80;
    deform_im2col_bf16<<<blocks_for(td), 256, 0, stream>>>(
        funr[i], offsets, Abuf, h, h, (uint32_t)td);
    // deformable conv contraction
    gemm(Abuf, dcn_wp, dcn_b, outp + outOff[i], P, 5, 16, h, 0);
  }
}
